// PaiNNMessage_16887811407945
// MI455X (gfx1250) — matrix operand order repaired and checked by reading, not compiled, not run on hardware
//
#include <hip/hip_runtime.h>
#include <hip/hip_bf16.h>
#include <hip/hip_fp16.h>

// ---------------------------------------------------------------------------
// PaiNN message layer for MI455X (gfx1250): wave32 + WMMA f32_16x16x32_f16.
// Per-node phi-MLP precompute (8x FLOP reduction vs per-edge), fragment-
// preswizzled f16 weights, LDS re-striping between WMMA layers, fp32 atomics
// for segment sums (L2-resident: s+v+phi_out ~ 180MB < 192MB L2).
// ---------------------------------------------------------------------------

typedef _Float16 half_t;
typedef __attribute__((ext_vector_type(16))) _Float16 v16h;
typedef __attribute__((ext_vector_type(8)))  _Float16 v8h;
typedef __attribute__((ext_vector_type(8)))  float    v8f;

#define Hdim   128
#define H3dim  384
#define NRBF   20
#define CUTOFF 8.0f

// fragment array offsets (in halves) inside d_ws
#define OFF_PHI_B1 0          // 4 kc * 8 tiles  * 512 = 16384
#define OFF_PHI_B2 16384      // 4 kc * 24 tiles * 512 = 49152
#define OFF_RBF_B1 65536      // 1 kc * 8 tiles  * 512 = 4096
#define OFF_RBF_B2 69632      // 4 kc * 24 tiles * 512 = 49152
#define TOTAL_FRAG_HALVES 118784
#define PHI_OUT_BYTE_OFF (256 * 1024)   // f32 phi_out [N][384] lives here

__device__ __forceinline__ float silu_f(float x) {
    return x * (1.0f / (1.0f + __expf(-x)));
}
__device__ __forceinline__ v8f bcast8(float x) {
    v8f r;
#pragma unroll
    for (int i = 0; i < 8; ++i) r[i] = x;
    return r;
}
__device__ __forceinline__ v8f wmma_f16(v16h a, v16h b, v8f c) {
    return __builtin_amdgcn_wmma_f32_16x16x32_f16(false, a, false, b,
                                                  (short)0, c, false, false);
}

// ---------------------------------------------------------------------------
// Kernel 1: pre-swizzle fp32 weights into f16 WMMA B-fragments.
// B-fragment lane map (16x16 tile, K-chunk of 32):
//   lane l: col n = 16*c + (l&15);  element e (0..15): k = 32*kc + 16*(e>>3) + 8*(l>>4) + (e&7)
// Stored contiguously per lane (32B) so the GEMM kernels load v16h directly.
// ---------------------------------------------------------------------------
__global__ __launch_bounds__(256) void convert_weights_kernel(
    const float* __restrict__ phi_w1, const float* __restrict__ phi_w2,
    const float* __restrict__ w_w1,   const float* __restrict__ w_w2,
    half_t* __restrict__ frags)
{
    int idx = blockIdx.x * 256 + threadIdx.x;
    if (idx >= TOTAL_FRAG_HALVES) return;
    const float* W; int C, Kreal, Kstride, rel;
    if (idx < OFF_PHI_B2)      { W = phi_w1; C = 8;  Kreal = 128;  Kstride = 128;  rel = idx - OFF_PHI_B1; }
    else if (idx < OFF_RBF_B1) { W = phi_w2; C = 24; Kreal = 128;  Kstride = 128;  rel = idx - OFF_PHI_B2; }
    else if (idx < OFF_RBF_B2) { W = w_w1;   C = 8;  Kreal = NRBF; Kstride = NRBF; rel = idx - OFF_RBF_B1; }
    else                       { W = w_w2;   C = 24; Kreal = 128;  Kstride = 128;  rel = idx - OFF_RBF_B2; }
    int frag = rel >> 9;          // /512 halves per fragment
    int lane = (rel >> 4) & 31;
    int e    = rel & 15;
    int kc = frag / C, c = frag % C;
    int k = 32 * kc + 16 * (e >> 3) + 8 * (lane >> 4) + (e & 7);
    int n = 16 * c + (lane & 15);
    float val = (k < Kreal) ? W[n * Kstride + k] : 0.0f;
    frags[idx] = (half_t)val;
}

// ---------------------------------------------------------------------------
// Kernel 2: initialize outputs with s and v (atomics accumulate on top).
// ---------------------------------------------------------------------------
__global__ __launch_bounds__(256) void init_out_kernel(
    const float* __restrict__ s, const float* __restrict__ v,
    float* __restrict__ out, int nS, int nV)
{
    int i = blockIdx.x * 256 + threadIdx.x;
    if (i < nS)            out[i] = s[i];
    else if (i < nS + nV)  out[i] = v[i - nS];
}

// ---------------------------------------------------------------------------
// Kernel 3: per-node phi MLP: phi_out = silu(s @ W1^T + b1) @ W2^T + b2.
// One wave = 16 nodes. LDS tile re-stripes C-layout -> A-layout between layers.
// A-fragment map: lane l, row = l&15, g = l>>4,
//   element e: k = 32*kc + 16*(e>>3) + 8*g + (e&7)
// ---------------------------------------------------------------------------
__global__ __launch_bounds__(256) void phi_node_kernel(
    const float* __restrict__ s, const float* __restrict__ b1,
    const float* __restrict__ b2, const half_t* __restrict__ frags,
    float* __restrict__ phi_out, int nNodes)
{
    __shared__ _Float16 shHid[8][16 * Hdim];
    const int lane = threadIdx.x & 31;
    const int wave = threadIdx.x >> 5;
    const int row  = lane & 15;
    const int g    = lane >> 4;
    const int nodeBase = blockIdx.x * 128 + wave * 16;
    const half_t* fragB1 = frags + OFF_PHI_B1;
    const half_t* fragB2 = frags + OFF_PHI_B2;

    int srow = nodeBase + row;
    if (srow >= nNodes) srow = nNodes - 1;
    const float* sr = s + (size_t)srow * Hdim;

    // A fragments of s (f32 -> f16), 4 K-chunks of 32
    v16h As[4];
#pragma unroll
    for (int kc = 0; kc < 4; ++kc) {
        v16h a;
#pragma unroll
        for (int e = 0; e < 16; ++e) {
            int k = 32 * kc + 16 * (e >> 3) + 8 * g + (e & 7);
            a[e] = (half_t)sr[k];
        }
        As[kc] = a;
    }

    // Layer 1: hidden[16,128] = silu(s @ W1^T + b1) -> LDS tile [m][k]
    _Float16* hid = shHid[wave];
#pragma unroll
    for (int c = 0; c < 8; ++c) {
        v8f acc = bcast8(b1[c * 16 + row]);
#pragma unroll
        for (int kc = 0; kc < 4; ++kc) {
            v16h bf = *(const v16h*)(fragB1 + (size_t)((kc * 8 + c) * 32 + lane) * 16);
            acc = wmma_f16(As[kc], bf, acc);
        }
#pragma unroll
        for (int r = 0; r < 8; ++r) {
            int m = 8 * g + r;
            hid[m * Hdim + c * 16 + row] = (half_t)silu_f(acc[r]);
        }
    }
    asm volatile("s_wait_dscnt 0" ::: "memory");

    // Re-stripe hidden into A-fragments for layer 2
    v16h A2[4];
#pragma unroll
    for (int kc = 0; kc < 4; ++kc) {
        v8h lo = *(const v8h*)(hid + row * Hdim + 32 * kc + 8 * g);
        v8h hi = *(const v8h*)(hid + row * Hdim + 32 * kc + 16 + 8 * g);
        v16h a;
#pragma unroll
        for (int i = 0; i < 8; ++i) { a[i] = lo[i]; a[i + 8] = hi[i]; }
        A2[kc] = a;
    }

    // Layer 2: out[16,384] -> phi_out
    for (int c = 0; c < 24; ++c) {
        v8f acc = bcast8(b2[c * 16 + row]);
#pragma unroll
        for (int kc = 0; kc < 4; ++kc) {
            v16h bf = *(const v16h*)(fragB2 + (size_t)((kc * 24 + c) * 32 + lane) * 16);
            acc = wmma_f16(A2[kc], bf, acc);
        }
#pragma unroll
        for (int r = 0; r < 8; ++r) {
            int node = nodeBase + 8 * g + r;
            if (node < nNodes)
                phi_out[(size_t)node * H3dim + c * 16 + row] = acc[r];
        }
    }
}

// ---------------------------------------------------------------------------
// Kernel 4: per-edge. One wave = 16 edges. Filter MLP via WMMA, multiply by
// gathered phi_out[src], fused atomic scatter for dS and dV1*v + dV2*u.
// ---------------------------------------------------------------------------
__global__ __launch_bounds__(256) void edge_kernel(
    const float* __restrict__ pos, const float* __restrict__ v_in,
    const float* __restrict__ rbf_c, const float* __restrict__ rbf_w,
    const float* __restrict__ b1, const float* __restrict__ b2,
    const half_t* __restrict__ frags, const float* __restrict__ phi_out,
    const int* __restrict__ edge_index,
    float* __restrict__ s_out, float* __restrict__ v_out, int nEdges)
{
    __shared__ _Float16 shHid[8][16 * Hdim];
    __shared__ int   shSrc[8][16];
    __shared__ int   shDst[8][16];
    __shared__ float shU[8][16][3];

    const int lane = threadIdx.x & 31;
    const int wave = threadIdx.x >> 5;
    const int row  = lane & 15;
    const int g    = lane >> 4;
    const int eBase = blockIdx.x * 128 + wave * 16;

    int e = eBase + row;
    if (e >= nEdges) e = nEdges - 1;
    int src = edge_index[e];
    int dst = edge_index[nEdges + e];

    float rx = pos[dst * 3 + 0] - pos[src * 3 + 0];
    float ry = pos[dst * 3 + 1] - pos[src * 3 + 1];
    float rz = pos[dst * 3 + 2] - pos[src * 3 + 2];
    float d  = fmaxf(sqrtf(rx * rx + ry * ry + rz * rz), 1e-6f);
    float id = 1.0f / d;
    float ux = rx * id, uy = ry * id, uz = rz * id;
    float xc = d * (1.0f / CUTOFF);
    float env = (xc < 1.0f) ? 0.5f * (__cosf(3.14159265358979f * xc) + 1.0f) : 0.0f;

    if (g == 0) {
        shSrc[wave][row] = src;
        shDst[wave][row] = dst;
        shU[wave][row][0] = ux; shU[wave][row][1] = uy; shU[wave][row][2] = uz;
    }

    // RBF features computed directly in A-fragment layout (K padded 20->32)
    v16h Af;
#pragma unroll
    for (int e2 = 0; e2 < 16; ++e2) {
        int k = 16 * (e2 >> 3) + 8 * g + (e2 & 7);
        float val = 0.0f;
        if (k < NRBF) {
            float dd = d - rbf_c[k];
            val = __expf(-fabsf(rbf_w[k]) * dd * dd) * env;
        }
        Af[e2] = (half_t)val;
    }

    // Layer 1: hidden = silu(feat @ Ww1^T + wb1) -> LDS
    _Float16* hid = shHid[wave];
    const half_t* fragR1 = frags + OFF_RBF_B1;
    const half_t* fragR2 = frags + OFF_RBF_B2;
#pragma unroll
    for (int c = 0; c < 8; ++c) {
        v8f acc = bcast8(b1[c * 16 + row]);
        v16h bf = *(const v16h*)(fragR1 + (size_t)(c * 32 + lane) * 16);
        acc = wmma_f16(Af, bf, acc);
#pragma unroll
        for (int r = 0; r < 8; ++r) {
            int m = 8 * g + r;
            hid[m * Hdim + c * 16 + row] = (half_t)silu_f(acc[r]);
        }
    }
    asm volatile("s_wait_dscnt 0" ::: "memory");

    v16h A2[4];
#pragma unroll
    for (int kc = 0; kc < 4; ++kc) {
        v8h lo = *(const v8h*)(hid + row * Hdim + 32 * kc + 8 * g);
        v8h hi = *(const v8h*)(hid + row * Hdim + 32 * kc + 16 + 8 * g);
        v16h a;
#pragma unroll
        for (int i = 0; i < 8; ++i) { a[i] = lo[i]; a[i + 8] = hi[i]; }
        A2[kc] = a;
    }

    // Per-lane metadata for the 8 C-rows (m = 8*g + r) this lane writes.
    int srcs[8], dsts[8];
    float uxr[8], uyr[8], uzr[8];
#pragma unroll
    for (int r = 0; r < 8; ++r) {
        int m = 8 * g + r;
        srcs[r] = shSrc[wave][m];
        dsts[r] = shDst[wave][m];
        uxr[r] = shU[wave][m][0];
        uyr[r] = shU[wave][m][1];
        uzr[r] = shU[wave][m][2];
    }

    // dS tiles (cols 0..127): scatter into s_out
    for (int c = 0; c < 8; ++c) {
        v8f acc = bcast8(b2[c * 16 + row]);
#pragma unroll
        for (int kc = 0; kc < 4; ++kc) {
            v16h bf = *(const v16h*)(fragR2 + (size_t)((kc * 24 + c) * 32 + lane) * 16);
            acc = wmma_f16(A2[kc], bf, acc);
        }
        int n = c * 16 + row;
#pragma unroll
        for (int r = 0; r < 8; ++r) {
            if (eBase + 8 * g + r < nEdges) {
                float val = acc[r] * phi_out[(size_t)srcs[r] * H3dim + n];
                atomicAdd(&s_out[(size_t)dsts[r] * Hdim + n], val);
            }
        }
    }

    // dV tiles: pair tile c (dV1, cols 128..255) with c+8 (dV2, cols 256..383)
    // so each v_out element takes exactly one fused atomic add.
    for (int c = 8; c < 16; ++c) {
        v8f acc1 = bcast8(b2[c * 16 + row]);
        v8f acc2 = bcast8(b2[(c + 8) * 16 + row]);
#pragma unroll
        for (int kc = 0; kc < 4; ++kc) {
            v16h bf1 = *(const v16h*)(fragR2 + (size_t)((kc * 24 + c) * 32 + lane) * 16);
            acc1 = wmma_f16(A2[kc], bf1, acc1);
            v16h bf2 = *(const v16h*)(fragR2 + (size_t)((kc * 24 + c + 8) * 32 + lane) * 16);
            acc2 = wmma_f16(A2[kc], bf2, acc2);
        }
        int h = (c - 8) * 16 + row;
#pragma unroll
        for (int r = 0; r < 8; ++r) {
            if (eBase + 8 * g + r < nEdges) {
                const float* ph = phi_out + (size_t)srcs[r] * H3dim;
                float dv1 = acc1[r] * ph[Hdim + h];
                float dv2 = acc2[r] * ph[2 * Hdim + h];
                const float* vv = v_in + ((size_t)srcs[r] * Hdim + h) * 3;
                float* vo = v_out + ((size_t)dsts[r] * Hdim + h) * 3;
                atomicAdd(&vo[0], dv1 * vv[0] + dv2 * uxr[r]);
                atomicAdd(&vo[1], dv1 * vv[1] + dv2 * uyr[r]);
                atomicAdd(&vo[2], dv1 * vv[2] + dv2 * uzr[r]);
            }
        }
    }
}

// ---------------------------------------------------------------------------
extern "C" void kernel_launch(void* const* d_in, const int* in_sizes, int n_in,
                              void* d_out, int out_size, void* d_ws, size_t ws_size,
                              hipStream_t stream)
{
    const float* s      = (const float*)d_in[0];
    const float* v      = (const float*)d_in[1];
    const float* pos    = (const float*)d_in[2];
    const float* rbf_c  = (const float*)d_in[3];
    const float* rbf_w  = (const float*)d_in[4];
    const float* phi_w1 = (const float*)d_in[5];
    const float* phi_b1 = (const float*)d_in[6];
    const float* phi_w2 = (const float*)d_in[7];
    const float* phi_b2 = (const float*)d_in[8];
    const float* w_w1   = (const float*)d_in[9];
    const float* w_b1   = (const float*)d_in[10];
    const float* w_w2   = (const float*)d_in[11];
    const float* w_b2   = (const float*)d_in[12];
    const int*   eidx   = (const int*)d_in[13];

    const int nNodes = in_sizes[0] / Hdim;
    const int nEdges = in_sizes[13] / 2;

    half_t* frags   = (half_t*)d_ws;
    float*  phi_out = (float*)((char*)d_ws + PHI_OUT_BYTE_OFF);
    float*  s_out   = (float*)d_out;
    float*  v_out   = s_out + (size_t)nNodes * Hdim;

    convert_weights_kernel<<<(TOTAL_FRAG_HALVES + 255) / 256, 256, 0, stream>>>(
        phi_w1, phi_w2, w_w1, w_w2, frags);

    int totOut = nNodes * Hdim + nNodes * Hdim * 3;
    init_out_kernel<<<(totOut + 255) / 256, 256, 0, stream>>>(
        s, v, s_out, nNodes * Hdim, nNodes * Hdim * 3);

    phi_node_kernel<<<(nNodes + 127) / 128, 256, 0, stream>>>(
        s, phi_b1, phi_b2, frags, phi_out, nNodes);

    edge_kernel<<<(nEdges + 127) / 128, 256, 0, stream>>>(
        pos, v, rbf_c, rbf_w, w_b1, w_b2, frags, phi_out, eidx,
        s_out, v_out, nEdges);

    (void)n_in; (void)out_size; (void)ws_size;
}
